// NexusNet_71914932404561
// MI455X (gfx1250) — compile-verified, matrix operand order checked by reading, not run on hardware
//
#include <hip/hip_runtime.h>

// Problem constants (match reference)
#define Nn   100000
#define Mm   30000
#define Ee   300000
#define Cc   5
#define PFf  64
#define NFf  32
#define Pp   3
#define CAT  192   // 3*PF  (nexus concat input)
#define HIN  96    // PF+NF (decoder concat input)

typedef __attribute__((ext_vector_type(16))) __bf16 v16bf;
typedef __attribute__((ext_vector_type(8)))  float  v8f;

// ---------------------------------------------------------------------------
// WMMA helpers (CDNA5 wave32, D = A(16x32 bf16) * B(32x16 bf16) + C(16x16 f32))
// ---------------------------------------------------------------------------
__device__ __forceinline__ v8f wmma_bf16(v16bf a, v16bf b, v8f c) {
  return __builtin_amdgcn_wmma_f32_16x16x32_bf16(false, a, false, b, (short)0, c,
                                                 false, false);
}

// 16-bit A/B operand layout (ISA 7.12.2):
//   lane L holds row (L&15); element i holds K = kbase + ((L>>4)<<3)
//                                             + ((i>>3)<<4) + (i&7)
__device__ __forceinline__ v16bf gather_bf(const __bf16* base, int lane,
                                           int stride, int row0, int kbase) {
  const __bf16* rp = base + (row0 + (lane & 15)) * stride;
  const int k0 = kbase + ((lane >> 4) << 3);
  v16bf r;
#pragma unroll
  for (int i = 0; i < 16; ++i)
    r[i] = rp[k0 + ((i >> 3) << 4) + (i & 7)];
  return r;
}

__device__ __forceinline__ v16bf gather_f32(const float* base, int lane,
                                            int stride, int kbase) {
  const float* rp = base + (lane & 15) * stride;
  const int k0 = kbase + ((lane >> 4) << 3);
  v16bf r;
#pragma unroll
  for (int i = 0; i < 16; ++i)
    r[i] = (__bf16)rp[k0 + ((i >> 3) << 4) + (i & 7)];
  return r;
}

// ---------------------------------------------------------------------------
// K0: zero a float region (grid-stride)
// ---------------------------------------------------------------------------
__global__ void k_zero(float* p, long long n) {
  long long i = (long long)blockIdx.x * blockDim.x + threadIdx.x;
  const long long s = (long long)gridDim.x * blockDim.x;
  for (; i < n; i += s) p[i] = 0.0f;
}

// ---------------------------------------------------------------------------
// K1: f32 -> bf16 weight conversion
// ---------------------------------------------------------------------------
__global__ void k_f2bf(const float* __restrict__ src, __bf16* __restrict__ dst,
                       int n) {
  int i = blockIdx.x * blockDim.x + threadIdx.x;
  const int s = gridDim.x * blockDim.x;
  for (; i < n; i += s) dst[i] = (__bf16)src[i];
}

// ---------------------------------------------------------------------------
// K2: scatter-add plane hits onto nexus accumulator
//     n_acc[tgt][c][plane*64 + f] += x[src][c][f]
//     one thread per (edge, class, 4-float chunk)
// ---------------------------------------------------------------------------
__global__ void k_scatter(const float* __restrict__ x, const int* __restrict__ ei,
                          float* __restrict__ nacc, int plane) {
  const long long total = (long long)Ee * Cc * (PFf / 4);
  long long t = (long long)blockIdx.x * blockDim.x + threadIdx.x;
  if (t >= total) return;
  const int e = (int)(t / (Cc * (PFf / 4)));
  const int r = (int)(t % (Cc * (PFf / 4)));
  const int c = r / (PFf / 4);
  const int f = (r % (PFf / 4)) * 4;
  const int src = ei[e];
  const int tgt = ei[Ee + e];
  const float4 v = *(const float4*)(x + ((long long)src * Cc + c) * PFf + f);
  float* d = nacc + ((long long)tgt * Cc + c) * CAT + plane * PFf + f;
  atomicAdd(d + 0, v.x);
  atomicAdd(d + 1, v.y);
  atomicAdd(d + 2, v.z);
  atomicAdd(d + 3, v.w);
}

// ---------------------------------------------------------------------------
// K3: nexus MLP  n = tanh(W2 * tanh(W1 * n_acc + b1) + b2)
//     grid (Mm/16/K3_TILES, Cc), block 64 (2 waves); wave w owns output
//     columns [16w,16w+16). Weights staged in LDS, B register tiles built once.
// ---------------------------------------------------------------------------
#define K3_TILES 5
__global__ void k_nexus_mlp(const float* __restrict__ nacc,
                            const __bf16* __restrict__ w1,
                            const __bf16* __restrict__ w2,
                            const float* __restrict__ b1,
                            const float* __restrict__ b2,
                            float* __restrict__ nout) {
  const int c = blockIdx.y;
  const int lane = threadIdx.x & 31;
  const int wv = threadIdx.x >> 5;  // 0..1

  __shared__ __align__(16) __bf16 sW1[NFf * CAT];  // 12 KB
  __shared__ __align__(16) __bf16 sW2[NFf * NFf];  // 2 KB
  __shared__ __align__(16) float  sN[16 * CAT];    // 12 KB (A source, f32)
  __shared__ __align__(16) __bf16 sH[16 * NFf];    // 1 KB  (layer-1 act)

  {  // cooperative weight staging (dword copies)
    const uint32_t* s1 = (const uint32_t*)(w1 + (long long)c * NFf * CAT);
    uint32_t* d1 = (uint32_t*)sW1;
    for (int i = threadIdx.x; i < NFf * CAT / 2; i += blockDim.x) d1[i] = s1[i];
    const uint32_t* s2 = (const uint32_t*)(w2 + (long long)c * NFf * NFf);
    uint32_t* d2 = (uint32_t*)sW2;
    for (int i = threadIdx.x; i < NFf * NFf / 2; i += blockDim.x) d2[i] = s2[i];
  }
  __syncthreads();

  // B register tiles: B[k][o] = W[o][k]  (Wt gather, built once per block)
  v16bf B1[CAT / 32];
#pragma unroll
  for (int ks = 0; ks < CAT / 32; ++ks)
    B1[ks] = gather_bf(sW1, lane, CAT, wv * 16, ks * 32);
  v16bf B2 = gather_bf(sW2, lane, NFf, wv * 16, 0);

  const int o = wv * 16 + (lane & 15);
  const float bias1 = b1[c * NFf + o];
  const float bias2 = b2[c * NFf + o];

  int m0 = blockIdx.x * K3_TILES * 16;
  for (int t = 0; t < K3_TILES; ++t, m0 += 16) {
    __syncthreads();
    // stage 16x192 f32 tile of n_acc
    for (int i = threadIdx.x; i < 16 * CAT / 4; i += blockDim.x) {
      const int row = i / (CAT / 4);
      const int f4 = (i % (CAT / 4)) * 4;
      *(float4*)(sN + row * CAT + f4) =
          *(const float4*)(nacc + ((long long)(m0 + row) * Cc + c) * CAT + f4);
    }
    __syncthreads();

    v8f acc;
#pragma unroll
    for (int r = 0; r < 8; ++r) acc[r] = bias1;
#pragma unroll
    for (int ks = 0; ks < CAT / 32; ++ks) {
      v16bf a = gather_f32(sN, lane, CAT, ks * 32);
      acc = wmma_bf16(a, B1[ks], acc);
    }
#pragma unroll
    for (int r = 0; r < 8; ++r) {
      const int Mr = r + ((lane >> 4) << 3);
      sH[Mr * NFf + o] = (__bf16)tanhf(acc[r]);
    }
    __syncthreads();

    v8f acc2;
#pragma unroll
    for (int r = 0; r < 8; ++r) acc2[r] = bias2;
    {
      v16bf a = gather_bf(sH, lane, NFf, 0, 0);
      acc2 = wmma_bf16(a, B2, acc2);
    }
#pragma unroll
    for (int r = 0; r < 8; ++r) {
      const int Mr = r + ((lane >> 4) << 3);
      nout[((long long)(m0 + Mr) * Cc + c) * NFf + o] = tanhf(acc2[r]);
    }
  }
}

// ---------------------------------------------------------------------------
// K4: edge attention + message mean-aggregation (per plane)
//     one thread per edge: 5-class logits, softmax over classes,
//     atomic scatter of att*n_j onto the source plane node, plus count.
// ---------------------------------------------------------------------------
__global__ void k_edge(const float* __restrict__ x, const float* __restrict__ n,
                       const int* __restrict__ ei,
                       const float* __restrict__ edW,
                       const float* __restrict__ edb,
                       float* __restrict__ agg, float* __restrict__ cnt) {
  __shared__ float sW[Cc * HIN];
  __shared__ float sB[Cc];
  for (int i = threadIdx.x; i < Cc * HIN; i += blockDim.x) sW[i] = edW[i];
  if (threadIdx.x < Cc) sB[threadIdx.x] = edb[threadIdx.x];
  __syncthreads();

  const int e = blockIdx.x * blockDim.x + threadIdx.x;
  if (e >= Ee) return;
  const int src = ei[e];
  const int tgt = ei[Ee + e];

  float logit[Cc];
#pragma unroll
  for (int c = 0; c < Cc; ++c) {
    const float* wrow = sW + c * HIN;
    const float* xp = x + ((long long)src * Cc + c) * PFf;
    const float* np = n + ((long long)tgt * Cc + c) * NFf;
    float s = sB[c];
#pragma unroll
    for (int f = 0; f < PFf; f += 4) {
      const float4 v = *(const float4*)(xp + f);
      s += wrow[f] * v.x + wrow[f + 1] * v.y + wrow[f + 2] * v.z +
           wrow[f + 3] * v.w;
    }
#pragma unroll
    for (int f = 0; f < NFf; f += 4) {
      const float4 v = *(const float4*)(np + f);
      s += wrow[PFf + f] * v.x + wrow[PFf + f + 1] * v.y +
           wrow[PFf + f + 2] * v.z + wrow[PFf + f + 3] * v.w;
    }
    logit[c] = s;
  }
  // softmax over the 5 classes
  float mx = logit[0];
#pragma unroll
  for (int c = 1; c < Cc; ++c) mx = fmaxf(mx, logit[c]);
  float sum = 0.0f;
  float att[Cc];
#pragma unroll
  for (int c = 0; c < Cc; ++c) {
    att[c] = __expf(logit[c] - mx);
    sum += att[c];
  }
  const float inv = 1.0f / sum;

  atomicAdd(&cnt[src], 1.0f);
#pragma unroll
  for (int c = 0; c < Cc; ++c) {
    const float a = att[c] * inv;
    const float* np = n + ((long long)tgt * Cc + c) * NFf;
    float* ap = agg + ((long long)src * Cc + c) * NFf;
#pragma unroll 4
    for (int f = 0; f < NFf; ++f) atomicAdd(ap + f, a * np[f]);
  }
}

// ---------------------------------------------------------------------------
// K5: node decoder  h = tanh(W2 * tanh(W1 * [x, agg/cnt] + b1) + b2)
//     grid (Nn/16/K5_TILES, Cc), block 128 (4 waves); wave w owns output
//     columns [16w,16w+16). Weights staged in LDS once; B register tiles
//     amortized over K5_TILES row tiles.
// ---------------------------------------------------------------------------
#define K5_TILES 10
__global__ void k_decoder(const float* __restrict__ x,
                          const float* __restrict__ agg,
                          const float* __restrict__ cnt,
                          const __bf16* __restrict__ w1,
                          const __bf16* __restrict__ w2,
                          const float* __restrict__ b1,
                          const float* __restrict__ b2,
                          float* __restrict__ out) {
  const int c = blockIdx.y;
  const int lane = threadIdx.x & 31;
  const int wv = threadIdx.x >> 5;  // 0..3

  __shared__ __align__(16) __bf16 sW1[PFf * HIN];  // 12 KB
  __shared__ __align__(16) __bf16 sW2[PFf * PFf];  // 8 KB
  __shared__ __align__(16) __bf16 sA[16 * HIN];    // 3 KB (layer-1 input)
  __shared__ __align__(16) __bf16 sH[16 * PFf];    // 2 KB (layer-1 act)

  {
    const uint32_t* s1 = (const uint32_t*)(w1 + (long long)c * PFf * HIN);
    uint32_t* d1 = (uint32_t*)sW1;
    for (int i = threadIdx.x; i < PFf * HIN / 2; i += blockDim.x) d1[i] = s1[i];
    const uint32_t* s2 = (const uint32_t*)(w2 + (long long)c * PFf * PFf);
    uint32_t* d2 = (uint32_t*)sW2;
    for (int i = threadIdx.x; i < PFf * PFf / 2; i += blockDim.x) d2[i] = s2[i];
  }
  __syncthreads();

  v16bf B1[HIN / 32];
#pragma unroll
  for (int ks = 0; ks < HIN / 32; ++ks)
    B1[ks] = gather_bf(sW1, lane, HIN, wv * 16, ks * 32);
  v16bf B2[PFf / 32];
#pragma unroll
  for (int ks = 0; ks < PFf / 32; ++ks)
    B2[ks] = gather_bf(sW2, lane, PFf, wv * 16, ks * 32);

  const int o = wv * 16 + (lane & 15);
  const float bias1 = b1[c * PFf + o];
  const float bias2 = b2[c * PFf + o];

  int n0 = blockIdx.x * K5_TILES * 16;
  for (int t = 0; t < K5_TILES; ++t, n0 += 16) {
    __syncthreads();
    // build 16x96 bf16 input tile: [x | agg/max(cnt,1)]
    for (int i = threadIdx.x; i < 16 * HIN; i += blockDim.x) {
      const int row = i / HIN;
      const int f = i % HIN;
      const long long nd = n0 + row;
      float v;
      if (f < PFf) {
        v = x[(nd * Cc + c) * PFf + f];
      } else {
        const float cc = fmaxf(cnt[nd], 1.0f);
        v = agg[(nd * Cc + c) * NFf + (f - PFf)] / cc;
      }
      sA[row * HIN + f] = (__bf16)v;
    }
    __syncthreads();

    v8f acc;
#pragma unroll
    for (int r = 0; r < 8; ++r) acc[r] = bias1;
#pragma unroll
    for (int ks = 0; ks < HIN / 32; ++ks) {
      v16bf a = gather_bf(sA, lane, HIN, 0, ks * 32);
      acc = wmma_bf16(a, B1[ks], acc);
    }
#pragma unroll
    for (int r = 0; r < 8; ++r) {
      const int Mr = r + ((lane >> 4) << 3);
      sH[Mr * PFf + o] = (__bf16)tanhf(acc[r]);
    }
    __syncthreads();

    v8f acc2;
#pragma unroll
    for (int r = 0; r < 8; ++r) acc2[r] = bias2;
#pragma unroll
    for (int ks = 0; ks < PFf / 32; ++ks) {
      v16bf a = gather_bf(sH, lane, PFf, 0, ks * 32);
      acc2 = wmma_bf16(a, B2[ks], acc2);
    }
#pragma unroll
    for (int r = 0; r < 8; ++r) {
      const int Mr = r + ((lane >> 4) << 3);
      out[((long long)(n0 + Mr) * Cc + c) * PFf + o] = tanhf(acc2[r]);
    }
  }
}

// ---------------------------------------------------------------------------
// Launch
// ---------------------------------------------------------------------------
extern "C" void kernel_launch(void* const* d_in, const int* in_sizes, int n_in,
                              void* d_out, int out_size, void* d_ws,
                              size_t ws_size, hipStream_t stream) {
  (void)in_sizes; (void)n_in; (void)out_size; (void)ws_size;
  // setup_inputs() dict order
  const float* x_u  = (const float*)d_in[0];
  const int*   ei_u = (const int*)d_in[1];
  const float* x_v  = (const float*)d_in[2];
  const int*   ei_v = (const int*)d_in[3];
  const float* x_y  = (const float*)d_in[4];
  const int*   ei_y = (const int*)d_in[5];
  // d_in[6] = nexus (zeros, shape-only)
  const float* nn_W1 = (const float*)d_in[7];
  const float* nn_b1 = (const float*)d_in[8];
  const float* nn_W2 = (const float*)d_in[9];
  const float* nn_b2 = (const float*)d_in[10];
  const float* ed_W  = (const float*)d_in[11];
  const float* ed_b  = (const float*)d_in[12];
  const float* nd_W1 = (const float*)d_in[13];
  const float* nd_b1 = (const float*)d_in[14];
  const float* nd_W2 = (const float*)d_in[15];
  const float* nd_b2 = (const float*)d_in[16];

  const float* xs[Pp]  = {x_u, x_v, x_y};
  const int*   eis[Pp] = {ei_u, ei_v, ei_y};

  char* ws = (char*)d_ws;
  // workspace layout (all offsets 256B-aligned)
  const long long SZ_NACC = (long long)Mm * Cc * CAT * 4;  // 115.2 MB
  const long long SZ_N    = (long long)Mm * Cc * NFf * 4;  //  19.2 MB
  const long long SZ_AGG  = (long long)Pp * Nn * Cc * NFf * 4;  // 192 MB
  const long long SZ_CNT  = (long long)Pp * Nn * 4;        //   1.2 MB
  long long off = 0;
  float* nacc = (float*)(ws + off); off += SZ_NACC;
  float* nf32 = (float*)(ws + off); off += SZ_N;
  float* aggAll = (float*)(ws + off); off += SZ_AGG;
  float* cntAll = (float*)(ws + off); off += SZ_CNT;
  off = (off + 255) & ~255LL;
  __bf16* w1bf  = (__bf16*)(ws + off); off += (long long)Cc * NFf * CAT * 2;
  __bf16* w2bf  = (__bf16*)(ws + off); off += (long long)Cc * NFf * NFf * 2;
  __bf16* nd1bf = (__bf16*)(ws + off); off += (long long)Pp * Cc * PFf * HIN * 2;
  __bf16* nd2bf = (__bf16*)(ws + off); off += (long long)Pp * Cc * PFf * PFf * 2;

  // 0) zero accumulators (n_acc, n, agg, cnt — contiguous)
  {
    const long long nz = (SZ_NACC + SZ_N + SZ_AGG + SZ_CNT) / 4;
    k_zero<<<2048, 256, 0, stream>>>((float*)ws, nz);
  }
  // 1) weights -> bf16
  k_f2bf<<<(Cc * NFf * CAT + 255) / 256, 256, 0, stream>>>(nn_W1, w1bf,
                                                           Cc * NFf * CAT);
  k_f2bf<<<(Cc * NFf * NFf + 255) / 256, 256, 0, stream>>>(nn_W2, w2bf,
                                                           Cc * NFf * NFf);
  k_f2bf<<<(Pp * Cc * PFf * HIN + 255) / 256, 256, 0, stream>>>(
      nd_W1, nd1bf, Pp * Cc * PFf * HIN);
  k_f2bf<<<(Pp * Cc * PFf * PFf + 255) / 256, 256, 0, stream>>>(
      nd_W2, nd2bf, Pp * Cc * PFf * PFf);

  // 2) scatter plane hits onto nexus accumulator
  {
    const long long tot = (long long)Ee * Cc * (PFf / 4);
    const int blocks = (int)((tot + 255) / 256);
    for (int p = 0; p < Pp; ++p)
      k_scatter<<<blocks, 256, 0, stream>>>(xs[p], eis[p], nacc, p);
  }

  // 3) nexus MLP (WMMA bf16)
  k_nexus_mlp<<<dim3(Mm / 16 / K3_TILES, Cc), 64, 0, stream>>>(
      nacc, w1bf, w2bf, nn_b1, nn_b2, nf32);

  // 4) edge attention + mean aggregation
  for (int p = 0; p < Pp; ++p) {
    k_edge<<<(Ee + 255) / 256, 256, 0, stream>>>(
        xs[p], nf32, eis[p], ed_W + (long long)p * Cc * HIN, ed_b + p * Cc,
        aggAll + (long long)p * Nn * Cc * NFf, cntAll + (long long)p * Nn);
  }

  // 5) node decoder (WMMA bf16), writes final outputs
  float* out = (float*)d_out;
  for (int p = 0; p < Pp; ++p) {
    k_decoder<<<dim3(Nn / 16 / K5_TILES, Cc), 128, 0, stream>>>(
        xs[p], aggAll + (long long)p * Nn * Cc * NFf,
        cntAll + (long long)p * Nn, nd1bf + (long long)p * Cc * PFf * HIN,
        nd2bf + (long long)p * Cc * PFf * PFf, nd_b1 + p * Cc * PFf,
        nd_b2 + p * Cc * PFf, out + (long long)p * Nn * Cc * PFf);
  }
}
